// SpikeSparseConnectomeAttention_10574209483452
// MI455X (gfx1250) — compile-verified, matrix operand order checked by reading, not run on hardware
//
#include <hip/hip_runtime.h>
#include <hip/hip_bf16.h>
#include <math.h>

// ---- problem constants ----
#define D_MODEL   512
#define N_HEADS   8
#define HEAD_DIM  64
#define N_CLUST   64
#define CLUST_KC  128
#define N_ROPE    32
#define BB        4
#define NN        8192

typedef _Float16 half_t;
typedef __attribute__((ext_vector_type(8)))  _Float16 v8h;
typedef __attribute__((ext_vector_type(16))) _Float16 v16h;
typedef __attribute__((ext_vector_type(8)))  float    v8f;

// ---- CDNA5 async global->LDS copy (ASYNCcnt path), with safe fallback ----
#if __has_builtin(__builtin_amdgcn_global_load_async_to_lds_b128)
#define HAVE_ASYNC_LDS 1
#else
#define HAVE_ASYNC_LDS 0
#endif

#if HAVE_ASYNC_LDS
typedef int v4i_ __attribute__((ext_vector_type(4)));
typedef __attribute__((address_space(1))) v4i_* as1_v4i;
typedef __attribute__((address_space(3))) v4i_* as3_v4i;
#endif

__device__ __forceinline__ void copy16_g2l(const half_t* g, half_t* l) {
#if HAVE_ASYNC_LDS
  // builtin signature: (v4i addrspace(1)*, v4i addrspace(3)*, imm offset, imm cpol)
  __builtin_amdgcn_global_load_async_to_lds_b128((as1_v4i)(size_t)g, (as3_v4i)l, 0, 0);
#else
  *(v8h*)l = *(const v8h*)g;
#endif
}

__device__ __forceinline__ void async_copy_wait() {
#if HAVE_ASYNC_LDS
  asm volatile("s_wait_asynccnt 0x0" ::: "memory");
#endif
}

// Load one 16x32-f16 WMMA operand fragment (A-style; also valid for B when the
// B matrix is K-contiguous per column). Per ISA layout: lanes 0-15 row m,
// halves {K0..7, K16..23}; lanes 16-31 same rows, halves {K8..15, K24..31}.
// Caller pre-offsets p by (lane>>4)*8.
__device__ __forceinline__ v16h load_frag16(const half_t* p) {
  v8h lo = *(const v8h*)(p);
  v8h hi = *(const v8h*)(p + 16);
  return __builtin_shufflevector(lo, hi, 0,1,2,3,4,5,6,7,8,9,10,11,12,13,14,15);
}

// ---------------- workspace layout (bytes) ----------------
constexpr size_t OFF_UPOS = 0;                       // B*N*3 f32
constexpr size_t OFF_XN   = 524288;                  // B*N*512 f16
constexpr size_t OFF_QK   = OFF_XN   + 33554432ull;
constexpr size_t OFF_WQ   = OFF_QK   + 33554432ull;  // 512*512 f16
constexpr size_t OFF_WK   = OFF_WQ   + 524288ull;
constexpr size_t OFF_WV   = OFF_WK   + 524288ull;
constexpr size_t OFF_Q    = OFF_WV   + 524288ull;    // B*N*512 f16
constexpr size_t OFF_K    = OFF_Q    + 33554432ull;
constexpr size_t OFF_V    = OFF_K    + 33554432ull;
constexpr size_t OFF_QINV = OFF_V    + 33554432ull;  // B*H*N f32
constexpr size_t OFF_KINV = OFF_QINV + 1048576ull;
constexpr size_t OFF_IIDX = OFF_KINV + 1048576ull;   // B*H*64*128 i32
constexpr size_t OFF_OIDX = OFF_IIDX + 1048576ull;

// ---------------- kernel 1: f32 -> f16 weight convert ----------------
__global__ void ssca_wconv(const float* __restrict__ qW, const float* __restrict__ kW,
                           const float* __restrict__ vW,
                           half_t* __restrict__ wq, half_t* __restrict__ wk,
                           half_t* __restrict__ wv) {
  int i = blockIdx.x * 256 + threadIdx.x;          // 0 .. 262143
  wq[i] = (half_t)qW[i];
  wk[i] = (half_t)kW[i];
  wv[i] = (half_t)vW[i];
}

// ---------------- kernel 2: RMSNorm + directional RoPE ----------------
__global__ void ssca_prep(const float* __restrict__ x, const float* __restrict__ ppos,
                          const float* __restrict__ rms_w, const float* __restrict__ rope_W,
                          const float* __restrict__ rdirs, const float* __restrict__ rfreq,
                          float* __restrict__ upos, half_t* __restrict__ xn_h,
                          half_t* __restrict__ qk_h) {
  int row = blockIdx.x;                  // b*NN + n
  int t = threadIdx.x;
  __shared__ float red[256];
  __shared__ float emb[2 * N_ROPE];
  __shared__ float up[3];
  const float* xr = x + (size_t)row * D_MODEL;
  float v0 = xr[2 * t], v1 = xr[2 * t + 1];
  red[t] = v0 * v0 + v1 * v1;
  __syncthreads();
  for (int st = 128; st > 0; st >>= 1) {
    if (t < st) red[t] += red[t + st];
    __syncthreads();
  }
  float rinv = rsqrtf(red[0] * (1.0f / D_MODEL) + 1e-6f);
  if (t == 0) {
    float px = ppos[3 * row], py = ppos[3 * row + 1], pz = ppos[3 * row + 2];
    float nrm = fmaxf(sqrtf(px * px + py * py + pz * pz), 1e-12f);
    up[0] = px / nrm; up[1] = py / nrm; up[2] = pz / nrm;
    upos[3 * row] = up[0]; upos[3 * row + 1] = up[1]; upos[3 * row + 2] = up[2];
  }
  __syncthreads();
  if (t < N_ROPE) {
    float a = (up[0] * rdirs[3 * t] + up[1] * rdirs[3 * t + 1] + up[2] * rdirs[3 * t + 2]) * rfreq[t];
    emb[t] = __sinf(a);
    emb[t + N_ROPE] = __cosf(a);
  }
  __syncthreads();
  #pragma unroll
  for (int j = 0; j < 2; ++j) {
    int d = 2 * t + j;
    const float* wrow = rope_W + (size_t)d * (2 * N_ROPE);
    float r = 0.f;
    #pragma unroll
    for (int f = 0; f < 2 * N_ROPE; ++f) r += emb[f] * wrow[f];
    float xnv = (j == 0 ? v0 : v1) * rinv * rms_w[d];
    xn_h[(size_t)row * D_MODEL + d] = (half_t)xnv;
    qk_h[(size_t)row * D_MODEL + d] = (half_t)(xnv + r);
  }
}

// ---------------- kernel 3: q/k/v projections via WMMA ----------------
// C[m,n] = sum_k A[m,k] * W[n,k]; block = 128(M) x 16(N), weight tile staged
// in LDS once (shared by all 8 waves); wave-per-16x16-tile.
#define WTS 520   // padded LDS row stride (halves): 520*2B=1040B, 16B aligned,
                  // 260 words -> 16 rows land on distinct banks
__global__ void ssca_proj(const half_t* __restrict__ qk, const half_t* __restrict__ xn,
                          const half_t* __restrict__ wq, const half_t* __restrict__ wk,
                          const half_t* __restrict__ wv,
                          half_t* __restrict__ qh, half_t* __restrict__ kh,
                          half_t* __restrict__ vh) {
  __shared__ half_t wt[16 * WTS];
  int which = blockIdx.z;
  const half_t* A = (which == 2) ? xn : qk;
  const half_t* W = (which == 0) ? wq : (which == 1 ? wk : wv);
  half_t* Dst     = (which == 0) ? qh : (which == 1 ? kh : vh);
  int n0 = blockIdx.y * 16;
  // stage W[n0..n0+15][0..511] -> LDS (1024 x 16B chunks, async if available)
  for (int i = threadIdx.x; i < 16 * 64; i += 256) {
    int r = i >> 6, c8 = (i & 63) * 8;
    copy16_g2l(W + (size_t)(n0 + r) * D_MODEL + c8, wt + r * WTS + c8);
  }
  async_copy_wait();
  __syncthreads();

  int wave = threadIdx.x >> 5, lane = threadIdx.x & 31;
  int lr = lane & 15, lh = lane >> 4;
  int m0 = blockIdx.x * 128 + wave * 16;
  const half_t* arow = A + (size_t)(m0 + lr) * D_MODEL + lh * 8;
  const half_t* brow = wt + lr * WTS + lh * 8;
  v8f acc = {};
  #pragma unroll 4
  for (int k0 = 0; k0 < D_MODEL; k0 += 32) {
    v16h a = load_frag16(arow + k0);
    v16h b = load_frag16(brow + k0);
    acc = __builtin_amdgcn_wmma_f32_16x16x32_f16(false, a, false, b, (short)0, acc, false, false);
  }
  int orow = m0 + lh * 8;
  int ocol = n0 + lr;
  #pragma unroll
  for (int r = 0; r < 8; ++r)
    Dst[(size_t)(orow + r) * D_MODEL + ocol] = (half_t)acc[r];
}

// ---------------- kernel 4: inverse norms of (q||pos), (k||pos) ----------------
__global__ void ssca_invnorm(const half_t* __restrict__ qh, const half_t* __restrict__ kh,
                             const float* __restrict__ upos,
                             float* __restrict__ qinv, float* __restrict__ kinv) {
  int idx = blockIdx.x * 256 + threadIdx.x;       // (b*H + h)*N + n
  int n = idx & (NN - 1);
  int b = idx / (N_HEADS * NN);
  const float* up = upos + ((size_t)b * NN + n) * 3;
  float ps = up[0] * up[0] + up[1] * up[1] + up[2] * up[2];
  const half_t* q = qh + (size_t)idx * HEAD_DIM;
  const half_t* k = kh + (size_t)idx * HEAD_DIM;
  float sq = 0.f, sk = 0.f;
  #pragma unroll
  for (int d8 = 0; d8 < 8; ++d8) {
    v8h a = *(const v8h*)(q + d8 * 8);
    v8h c = *(const v8h*)(k + d8 * 8);
    #pragma unroll
    for (int j = 0; j < 8; ++j) {
      float fa = (float)a[j]; sq += fa * fa;
      float fc = (float)c[j]; sk += fc * fc;
    }
  }
  qinv[idx] = 1.0f / fmaxf(sqrtf(sq + ps), 1e-12f);
  kinv[idx] = 1.0f / fmaxf(sqrtf(sk + ps), 1e-12f);
}

// ---------------- kernel 5: centroid score + top-128 selection ----------------
__global__ void ssca_score_topk(const half_t* __restrict__ qh, const half_t* __restrict__ kh,
                                const float* __restrict__ upos,
                                const float* __restrict__ qinvA, const float* __restrict__ kinvA,
                                const float* __restrict__ ic, const float* __restrict__ oc,
                                int* __restrict__ in_idx, int* __restrict__ out_idx) {
  int c = blockIdx.x, h = blockIdx.y;
  int b = blockIdx.z >> 1, which = blockIdx.z & 1;
  const half_t* feat = which ? kh : qh;
  const float* cen   = (which ? oc : ic) + ((size_t)h * N_CLUST + c) * (HEAD_DIM + 3);
  const float* finv  = which ? kinvA : qinvA;
  int* outp = (which ? out_idx : in_idx) + ((size_t)(b * N_HEADS + h) * N_CLUST + c) * CLUST_KC;
  __shared__ float sc[NN];
  __shared__ float cn[HEAD_DIM + 4];
  __shared__ float cinv_s;
  __shared__ float rv[256];
  __shared__ int   ri[256];
  int t = threadIdx.x;
  if (t < HEAD_DIM + 3) cn[t] = cen[t];
  __syncthreads();
  if (t == 0) {
    float s = 0.f;
    for (int j = 0; j < HEAD_DIM + 3; ++j) s += cn[j] * cn[j];
    cinv_s = 1.0f / fmaxf(sqrtf(s), 1e-12f);
  }
  __syncthreads();
  float cinv = cinv_s;
  size_t base = (size_t)(b * N_HEADS + h) * NN;
  for (int n = t; n < NN; n += 256) {
    const half_t* f = feat + (base + n) * HEAD_DIM;
    __builtin_prefetch(f + 256 * HEAD_DIM, 0, 1);   // -> global_prefetch_b8
    float s = 0.f;
    #pragma unroll
    for (int d8 = 0; d8 < 8; ++d8) {
      v8h a = *(const v8h*)(f + d8 * 8);
      #pragma unroll
      for (int j = 0; j < 8; ++j) s += (float)a[j] * cn[d8 * 8 + j];
    }
    const float* up = upos + ((size_t)b * NN + n) * 3;
    s += up[0] * cn[64] + up[1] * cn[65] + up[2] * cn[66];
    sc[n] = s * cinv * finv[base + n];
  }
  __syncthreads();
  for (int it = 0; it < CLUST_KC; ++it) {
    float bv = -1e30f; int bi = 0;
    for (int n = t; n < NN; n += 256) {
      float v = sc[n];
      if (v > bv) { bv = v; bi = n; }
    }
    rv[t] = bv; ri[t] = bi;
    __syncthreads();
    for (int st = 128; st > 0; st >>= 1) {
      if (t < st && rv[t + st] > rv[t]) { rv[t] = rv[t + st]; ri[t] = ri[t + st]; }
      __syncthreads();
    }
    if (t == 0) { outp[it] = ri[0]; sc[ri[0]] = -1e30f; }
    __syncthreads();
  }
}

// ---------------- kernel 6: per-cluster attention (WMMA x2 + softmax) ----------------
// padded LDS strides (halves / floats); all fragment bases stay 16B-aligned
#define QCS 80      // qc/kc row stride (halves)   160B -> distinct banks
#define VTS 136     // vT row stride  (halves)     272B
#define ATS 136     // attn row stride (halves)    272B
#define LGS 132     // logits row stride (floats)  528B
// dynamic LDS layout (bytes)
#define SM_QC    0            // 128*QCS*2 = 20480   -- reused for attn later
#define SM_KC    20480        // 20480
#define SM_ATTN  0            // 128*ATS*2 = 34816 (overlays qc+kc: 40960)
#define SM_VT    40960        // 64*VTS*2 = 17408
#define SM_LOG   58368        // 128*LGS*4 = 67584
#define SM_IIDX  125952       // 128 i32
#define SM_OIDX  126464       // 128 i32
#define SM_MADD  126976       // 128 f32
#define SM_FLAG  127488       // i32
#define SMEM_BYTES 127744

__global__ void ssca_attn(const half_t* __restrict__ qh, const half_t* __restrict__ kh,
                          const half_t* __restrict__ vh,
                          const int* __restrict__ in_idx, const int* __restrict__ out_idx,
                          const unsigned char* __restrict__ spike,
                          float* __restrict__ out) {
  extern __shared__ char smem[];
  half_t* qc   = (half_t*)(smem + SM_QC);
  half_t* kc   = (half_t*)(smem + SM_KC);
  half_t* attn = (half_t*)(smem + SM_ATTN);
  half_t* vT   = (half_t*)(smem + SM_VT);
  float*  lg   = (float*) (smem + SM_LOG);
  int*   iidx  = (int*)   (smem + SM_IIDX);
  int*   oidx  = (int*)   (smem + SM_OIDX);
  float* madd  = (float*) (smem + SM_MADD);
  int*   flag  = (int*)   (smem + SM_FLAG);

  int c = blockIdx.x, h = blockIdx.y, b = blockIdx.z;
  int t = threadIdx.x;
  size_t cl = (size_t)(b * N_HEADS + h) * N_CLUST + c;
  const int* gi = in_idx  + cl * CLUST_KC;
  const int* go = out_idx + cl * CLUST_KC;
  if (t == 0) *flag = 0;
  __syncthreads();
  if (t < CLUST_KC) {
    iidx[t] = gi[t];
    int no = go[t];
    oidx[t] = no;
    int sp = spike[(size_t)b * NN + no];
    madd[t] = sp ? 0.0f : -1e9f;
    if (sp) atomicOr(flag, 1);
  }
  __syncthreads();
  float oscale = (*flag) ? 1.0f : 0.0f;

  size_t base = (size_t)(b * N_HEADS + h) * NN * HEAD_DIM;
  for (int i = t; i < CLUST_KC * 8; i += 256) {         // 128 rows x 8 chunks
    int r = i >> 3, d8 = (i & 7) * 8;
    copy16_g2l(qh + base + (size_t)iidx[r] * HEAD_DIM + d8, qc + r * QCS + d8);
    copy16_g2l(kh + base + (size_t)oidx[r] * HEAD_DIM + d8, kc + r * QCS + d8);
    v8h vv = *(const v8h*)(vh + base + (size_t)oidx[r] * HEAD_DIM + d8);
    #pragma unroll
    for (int j = 0; j < 8; ++j) vT[(d8 + j) * VTS + r] = vv[j];
  }
  async_copy_wait();
  __syncthreads();

  int wave = t >> 5, lane = t & 31, lr = lane & 15, lh = lane >> 4;
  int m0 = wave * 16;

  // logits = (qc * kc^T) / sqrt(64) + mask
  const half_t* ar = qc + (m0 + lr) * QCS + lh * 8;
  #pragma unroll
  for (int nb = 0; nb < 8; ++nb) {
    const half_t* br = kc + (nb * 16 + lr) * QCS + lh * 8;
    v8f acc = {};
    #pragma unroll
    for (int kk = 0; kk < HEAD_DIM; kk += 32) {
      v16h a  = load_frag16(ar + kk);
      v16h bm = load_frag16(br + kk);
      acc = __builtin_amdgcn_wmma_f32_16x16x32_f16(false, a, false, bm, (short)0, acc, false, false);
    }
    int orow = m0 + lh * 8, ocol = nb * 16 + lr;
    #pragma unroll
    for (int r = 0; r < 8; ++r)
      lg[(orow + r) * LGS + ocol] = acc[r] * 0.125f + madd[ocol];
  }
  __syncthreads();

  // row softmax -> attn (f16, overlays qc/kc)
  if (t < CLUST_KC) {
    float* row = lg + t * LGS;
    float m = -1e30f;
    for (int k = 0; k < CLUST_KC; ++k) m = fmaxf(m, row[k]);
    float s = 0.f;
    for (int k = 0; k < CLUST_KC; ++k) s += __expf(row[k] - m);
    float inv = 1.0f / s;
    for (int k = 0; k < CLUST_KC; ++k)
      attn[t * ATS + k] = (half_t)(__expf(row[k] - m) * inv);
  }
  __syncthreads();

  // out = attn * vc  (vc transposed in LDS -> contiguous B fragments)
  const half_t* ar2 = attn + (m0 + lr) * ATS + lh * 8;
  float* outp = out + cl * CLUST_KC * HEAD_DIM;
  #pragma unroll
  for (int nb = 0; nb < 4; ++nb) {
    const half_t* br2 = vT + (nb * 16 + lr) * VTS + lh * 8;
    v8f acc = {};
    #pragma unroll
    for (int kk = 0; kk < CLUST_KC; kk += 32) {
      v16h a  = load_frag16(ar2 + kk);
      v16h bm = load_frag16(br2 + kk);
      acc = __builtin_amdgcn_wmma_f32_16x16x32_f16(false, a, false, bm, (short)0, acc, false, false);
    }
    int orow = m0 + lh * 8, ocol = nb * 16 + lr;
    #pragma unroll
    for (int r = 0; r < 8; ++r)
      outp[(size_t)(orow + r) * HEAD_DIM + ocol] = acc[r] * oscale;
  }
}

// ---------------- launch ----------------
extern "C" void kernel_launch(void* const* d_in, const int* in_sizes, int n_in,
                              void* d_out, int out_size, void* d_ws, size_t ws_size,
                              hipStream_t stream) {
  (void)in_sizes; (void)n_in; (void)out_size; (void)ws_size;
  const float* x      = (const float*)d_in[0];
  const float* ppos   = (const float*)d_in[1];
  const unsigned char* spike = (const unsigned char*)d_in[3];   // [B,1,N] bool
  const float* rms_w  = (const float*)d_in[5];
  const float* rope_W = (const float*)d_in[6];
  const float* qW     = (const float*)d_in[7];
  const float* kW     = (const float*)d_in[8];
  const float* vW     = (const float*)d_in[9];
  const float* rdirs  = (const float*)d_in[10];
  const float* rfreq  = (const float*)d_in[11];
  const float* ic     = (const float*)d_in[12];
  const float* oc     = (const float*)d_in[13];

  char* ws = (char*)d_ws;
  float*  upos = (float*) (ws + OFF_UPOS);
  half_t* xn_h = (half_t*)(ws + OFF_XN);
  half_t* qk_h = (half_t*)(ws + OFF_QK);
  half_t* wq   = (half_t*)(ws + OFF_WQ);
  half_t* wk   = (half_t*)(ws + OFF_WK);
  half_t* wv   = (half_t*)(ws + OFF_WV);
  half_t* q_h  = (half_t*)(ws + OFF_Q);
  half_t* k_h  = (half_t*)(ws + OFF_K);
  half_t* v_h  = (half_t*)(ws + OFF_V);
  float*  qinv = (float*) (ws + OFF_QINV);
  float*  kinv = (float*) (ws + OFF_KINV);
  int*    iidx = (int*)   (ws + OFF_IIDX);
  int*    oidx = (int*)   (ws + OFF_OIDX);

  ssca_wconv<<<(D_MODEL * D_MODEL) / 256, 256, 0, stream>>>(qW, kW, vW, wq, wk, wv);
  ssca_prep<<<BB * NN, 256, 0, stream>>>(x, ppos, rms_w, rope_W, rdirs, rfreq, upos, xn_h, qk_h);
  ssca_proj<<<dim3((BB * NN) / 128, D_MODEL / 16, 3), 256, 0, stream>>>(qk_h, xn_h, wq, wk, wv, q_h, k_h, v_h);
  ssca_invnorm<<<(BB * N_HEADS * NN) / 256, 256, 0, stream>>>(q_h, k_h, upos, qinv, kinv);
  ssca_score_topk<<<dim3(N_CLUST, N_HEADS, BB * 2), 256, 0, stream>>>(q_h, k_h, upos, qinv, kinv, ic, oc, iidx, oidx);
  ssca_attn<<<dim3(N_CLUST, N_HEADS, BB), 256, SMEM_BYTES, stream>>>(q_h, k_h, v_h, iidx, oidx, spike, (float*)d_out);
}
